// AttentionBlock_63445256896550
// MI455X (gfx1250) — matrix operand order repaired and checked by reading, not compiled, not run on hardware
//
#include <hip/hip_runtime.h>

// ---------------------------------------------------------------------------
// MI455X (gfx1250, wave32) attention block with LRP gamma-rule pproc.
// All GEMMs use v_wmma_f32_16x16x32_bf16 with fp32 accumulation.
// B operands staged TRANSPOSED in LDS -> fragment loads are 2x ds_load_b128.
// All staging loops are compile-time unrolled so global loads batch up
// (no per-iteration s_wait_loadcnt serialization).
// ---------------------------------------------------------------------------

typedef __bf16 bf16_t;
typedef __bf16 v16bf __attribute__((ext_vector_type(16)));
typedef float  v8f   __attribute__((ext_vector_type(8)));
typedef unsigned int u32x4 __attribute__((ext_vector_type(4)));

#define BB   2
#define SS   2048
#define HID  1024
#define NH   16
#define HD   64
#define MROWS (BB*SS)          // 4096
#define EPS_P  1e-5f
#define EPS_LN 1e-12f

// ---------------------------------------------------------------------------
// WMMA helpers (CDNA5 VGPR layouts, ISA 7.12.2)
// ---------------------------------------------------------------------------
__device__ __forceinline__ v8f wmma_bf16(v16bf a, v16bf b, v8f c) {
  return __builtin_amdgcn_wmma_f32_16x16x32_bf16(false, a, false, b, (short)0, c,
                                                 false, false);
}

// A fragment: 16x32 bf16 tile, row-major in LDS (ld elements per row).
// lane<16 : row M=lane,     elems 0..7 = K 0..7,  elems 8..15 = K 16..23
// lane>=16: row M=lane-16,  elems 0..7 = K 8..15, elems 8..15 = K 24..31
__device__ __forceinline__ v16bf load_frag_a(const bf16_t* tile, int ld, int lane) {
  int m  = lane & 15;
  int kb = (lane < 16) ? 0 : 8;
  v16bf a;
#pragma unroll
  for (int e = 0; e < 8; ++e) a[e]     = tile[m * ld + kb + e];
#pragma unroll
  for (int e = 0; e < 8; ++e) a[8 + e] = tile[m * ld + kb + 16 + e];
  return a;
}

// B fragment from an N-major (transposed) tile: tileT[n*ld + k].
// lane<16: col N=lane, K 0..7,16..23 ; lane>=16: col N=lane-16, K 8..15,24..31.
__device__ __forceinline__ v16bf load_frag_bt(const bf16_t* tileT, int ld, int lane) {
  int n  = lane & 15;
  int kb = (lane < 16) ? 0 : 8;
  v16bf b;
#pragma unroll
  for (int e = 0; e < 8; ++e) b[e] = tileT[n * ld + kb + e];
#pragma unroll
  for (int e = 0; e < 8; ++e) b[8 + e] = tileT[n * ld + kb + 16 + e];
  return b;
}

// ---------------------------------------------------------------------------
// Kernel 1: fp32 -> bf16 conversion (+ gamma-perturbed variant)
// ---------------------------------------------------------------------------
__global__ __launch_bounds__(256) void prep_convert(
    const float* __restrict__ src, bf16_t* __restrict__ dst,
    bf16_t* __restrict__ dstp, const float* __restrict__ gamma_p, int n) {
  float g = *gamma_p;
  for (int i = blockIdx.x * blockDim.x + threadIdx.x; i < n;
       i += gridDim.x * blockDim.x) {
    float v = src[i];
    if (dst)  dst[i]  = (bf16_t)v;
    if (dstp) dstp[i] = (bf16_t)(v + g * fmaxf(v, 0.f));
  }
}

// ---------------------------------------------------------------------------
// Kernel 2: QKV projection.  out = (zp+eps) * (z / (zp+eps))
//   z  = x @ W  + b ,  zp = x @ Wp + bp    (both GEMMs share the A operand)
// Block tile 128x64, 8 waves in 4x2 grid, each wave 32x32 (2x2 WMMA subtiles).
// ---------------------------------------------------------------------------
__global__ __launch_bounds__(256) void qkv_gemm(
    const bf16_t* __restrict__ xb,   // [4096,1024] bf16
    const bf16_t* __restrict__ Wb,   // [1024,1024] bf16 (plain)
    const bf16_t* __restrict__ Wpb,  // [1024,1024] bf16 (perturbed)
    const float*  __restrict__ bias, // [1024] fp32
    const float*  __restrict__ gamma_p,
    bf16_t* __restrict__ out)        // [4096,1024] bf16
{
  const int LDA = 40, LDK = 40;
  __shared__ __align__(16) bf16_t As[128 * LDA];
  __shared__ __align__(16) bf16_t Bzt[64 * LDK];
  __shared__ __align__(16) bf16_t Bpt[64 * LDK];

  int m0 = blockIdx.x * 128, n0 = blockIdx.y * 64;
  int tid = threadIdx.x, lane = tid & 31, w = tid >> 5;
  int wm = w & 3, wn = w >> 2;

  v8f accz[2][2] = {{{0}}}, accp[2][2] = {{{0}}};

  for (int k0 = 0; k0 < HID; k0 += 32) {
    // Stage A (128x32 row-major): 512 x 16B copies, 2 per thread, unrolled.
#pragma unroll
    for (int it = 0; it < 2; ++it) {
      int t = tid + it * 256;
      int r = t >> 2, c8 = (t & 3) * 8;
      *(u32x4*)&As[r * LDA + c8] =
          *(const u32x4*)&xb[(size_t)(m0 + r) * HID + k0 + c8];
    }
    // Stage both W tiles TRANSPOSED: Bt[n][k], 2 packed b64 stores/thread.
#pragma unroll
    for (int it = 0; it < 2; ++it) {
      int t = tid + it * 256;
      int n = t & 63, k4 = (t >> 6) * 4;
      union { unsigned long long u; bf16_t h[4]; } pz, pp;
#pragma unroll
      for (int j = 0; j < 4; ++j) {
        pz.h[j] = Wb [(size_t)(k0 + k4 + j) * HID + n0 + n];
        pp.h[j] = Wpb[(size_t)(k0 + k4 + j) * HID + n0 + n];
      }
      *(unsigned long long*)&Bzt[n * LDK + k4] = pz.u;
      *(unsigned long long*)&Bpt[n * LDK + k4] = pp.u;
    }
    if (k0 + 32 < HID) {
      __builtin_prefetch(&xb[(size_t)(m0 + (tid >> 1)) * HID + k0 + 32], 0, 0);
      __builtin_prefetch(&Wb[(size_t)(k0 + 32 + (tid & 31)) * HID + n0], 0, 0);
    }
    __syncthreads();

    v16bf afr[2];
    afr[0] = load_frag_a(&As[(wm * 32 + 0)  * LDA], LDA, lane);
    afr[1] = load_frag_a(&As[(wm * 32 + 16) * LDA], LDA, lane);
#pragma unroll
    for (int ni = 0; ni < 2; ++ni) {
      v16bf bz = load_frag_bt(&Bzt[(wn * 32 + ni * 16) * LDK], LDK, lane);
      v16bf bp = load_frag_bt(&Bpt[(wn * 32 + ni * 16) * LDK], LDK, lane);
#pragma unroll
      for (int mi = 0; mi < 2; ++mi) {
        accz[mi][ni] = wmma_bf16(afr[mi], bz, accz[mi][ni]);
        accp[mi][ni] = wmma_bf16(afr[mi], bp, accp[mi][ni]);
      }
    }
    __syncthreads();
  }

  float g = *gamma_p;
  int nlane = lane & 15, mb = (lane < 16) ? 0 : 8;
#pragma unroll
  for (int mi = 0; mi < 2; ++mi)
#pragma unroll
    for (int ni = 0; ni < 2; ++ni)
#pragma unroll
      for (int r = 0; r < 8; ++r) {
        int row = m0 + wm * 32 + mi * 16 + mb + r;
        int col = n0 + wn * 32 + ni * 16 + nlane;
        float bv  = bias[col];
        float bpv = bv + g * fmaxf(bv, 0.f);
        float z   = accz[mi][ni][r] + bv;
        float zp  = accp[mi][ni][r] + bpv;
        float val = (zp + EPS_P) * (z / (zp + EPS_P));
        out[(size_t)row * HID + col] = (bf16_t)val;
      }
}

// ---------------------------------------------------------------------------
// Kernel 3a: scores = q @ k^T (no scaling). Writes RAW scores into the probs
// output region and per-row (max, sum-exp) stats (two-pass softmax, pass 1).
// One block per (b, h, 16 query rows); 8 waves each own 1/8 of key columns.
// ---------------------------------------------------------------------------
__global__ __launch_bounds__(256) void attn_scores(
    const bf16_t* __restrict__ qb,   // [4096,1024] bf16
    const bf16_t* __restrict__ kb,   // [4096,1024] bf16
    float* __restrict__ probs,       // [B,NH,S,S] raw scores out
    float* __restrict__ rowstats)    // [B*NH*S][2]  (max, sumexp)
{
  const int LQ = 72;
  __shared__ __align__(16) bf16_t qs[16 * LQ];
  __shared__ __align__(16) bf16_t ks[128 * LQ];
  __shared__ float stats[8][16][2];

  int bid = blockIdx.x;
  int mt = bid & 127, bh = bid >> 7;     // S/16 = 128 row tiles
  int h = bh & 15, b = bh >> 4;
  int i0 = mt * 16;
  int tid = threadIdx.x, lane = tid & 31, w = tid >> 5;

  const bf16_t* qbase = qb + ((size_t)(b * SS + i0)) * HID + h * HD;
  if (tid < 64) {                        // 16 rows x 4 chunks of 8 elems
    int r = tid >> 2, c8 = (tid & 3) * 8;
    *(u32x4*)&qs[r * LQ + c8] =
        *(const u32x4*)&qbase[(size_t)r * HID + c8];
  }

  float mrun[8], lrun[8];
#pragma unroll
  for (int r = 0; r < 8; ++r) { mrun[r] = -__builtin_inff(); lrun[r] = 0.f; }

  int nlane = lane & 15, mb = (lane < 16) ? 0 : 8;

  for (int ch = 0; ch < 16; ++ch) {
    __syncthreads();
    int j0c = ch * 128;
    const bf16_t* kbase = kb + ((size_t)(b * SS + j0c)) * HID + h * HD;
#pragma unroll
    for (int it = 0; it < 2; ++it) {     // 128 rows x 4 chunks -> 2/thread
      int t = tid + it * 256;
      int r = t >> 2, c8 = (t & 3) * 8;
      *(u32x4*)&ks[r * LQ + c8] =
          *(const u32x4*)&kbase[(size_t)r * HID + c8];
    }
    __syncthreads();

    v8f acc = {0};
    v16bf a0 = load_frag_a(qs,      LQ, lane);   // K 0..31 of head dim
    v16bf a1 = load_frag_a(qs + 32, LQ, lane);   // K 32..63
    v16bf b0 = load_frag_bt(&ks[(w * 16) * LQ],      LQ, lane);
    v16bf b1 = load_frag_bt(&ks[(w * 16) * LQ + 32], LQ, lane);
    acc = wmma_bf16(a0, b0, acc);
    acc = wmma_bf16(a1, b1, acc);

    int j = j0c + w * 16 + nlane;
#pragma unroll
    for (int r = 0; r < 8; ++r) {
      float s = acc[r];
      int i = i0 + mb + r;
      probs[(((size_t)(b * NH + h) * SS) + i) * SS + j] = s;
      float mx = s;
      mx = fmaxf(mx, __shfl_xor(mx, 1, 32));
      mx = fmaxf(mx, __shfl_xor(mx, 2, 32));
      mx = fmaxf(mx, __shfl_xor(mx, 4, 32));
      mx = fmaxf(mx, __shfl_xor(mx, 8, 32));
      float mnew = fmaxf(mrun[r], mx);
      float pe = __expf(s - mnew);
      pe += __shfl_xor(pe, 1, 32);
      pe += __shfl_xor(pe, 2, 32);
      pe += __shfl_xor(pe, 4, 32);
      pe += __shfl_xor(pe, 8, 32);
      lrun[r] = lrun[r] * __expf(mrun[r] - mnew) + pe;
      mrun[r] = mnew;
    }
  }

  if ((lane & 15) == 0) {
#pragma unroll
    for (int r = 0; r < 8; ++r) {
      stats[w][mb + r][0] = mrun[r];
      stats[w][mb + r][1] = lrun[r];
    }
  }
  __syncthreads();
  if (tid < 16) {
    float M = -__builtin_inff(), L = 0.f;
    for (int ww = 0; ww < 8; ++ww) {
      float m2 = stats[ww][tid][0], l2 = stats[ww][tid][1];
      float Mn = fmaxf(M, m2);
      L = L * __expf(M - Mn) + l2 * __expf(m2 - Mn);
      M = Mn;
    }
    size_t rowid = (size_t)(b * NH + h) * SS + i0 + tid;
    rowstats[rowid * 2 + 0] = M;
    rowstats[rowid * 2 + 1] = L;
  }
}

// ---------------------------------------------------------------------------
// Kernel 3b: softmax pass 2 + ctx = probs @ v.
// Raw scores re-read in WMMA A-fragment layout, normalized, written back as
// final probs, and used directly as the A operand of P @ V.
// ---------------------------------------------------------------------------
__global__ __launch_bounds__(256) void attn_ctx(
    const bf16_t* __restrict__ vb,   // [4096,1024] bf16
    float* __restrict__ probs,       // raw in -> normalized out
    const float* __restrict__ rowstats,
    bf16_t* __restrict__ ctx)        // [4096,1024] bf16
{
  const int LDJ = 264;               // 256 keys + pad
  __shared__ __align__(16) bf16_t vst[64 * LDJ];   // 33792 B, reused for red
  __shared__ float rs[16][2];

  int bid = blockIdx.x;
  int mt = bid & 127, bh = bid >> 7;
  int h = bh & 15, b = bh >> 4;
  int i0 = mt * 16;
  int tid = threadIdx.x, lane = tid & 31, w = tid >> 5;

  if (tid < 16) {
    size_t rowid = (size_t)(b * NH + h) * SS + i0 + tid;
    rs[tid][0] = rowstats[rowid * 2 + 0];
    rs[tid][1] = rowstats[rowid * 2 + 1];
  }

  v8f O[4] = {{0}};
  int m = lane & 15;                 // A-fragment row for this lane
  int kb0 = (lane < 16) ? 0 : 8;

  for (int ch = 0; ch < 8; ++ch) {
    __syncthreads();                 // protects rs (ch=0) and vst reuse
    int j0c = ch * 256;
    const bf16_t* vbase = vb + ((size_t)(b * SS + j0c)) * HID + h * HD;
    // Stage V transposed: vst[d][j]; 16 packed b64 stores per thread.
#pragma unroll 4
    for (int it = 0; it < 16; ++it) {
      int t = tid + it * 256;
      int d = t & 63, j4 = (t >> 6) * 4;
      union { unsigned long long u; bf16_t h4[4]; } pk;
#pragma unroll
      for (int jj = 0; jj < 4; ++jj)
        pk.h4[jj] = vbase[(size_t)(j4 + jj) * HID + d];
      *(unsigned long long*)&vst[d * LDJ + j4] = pk.u;
    }
    __syncthreads();

    int j0 = j0c + w * 32;           // this wave's 32 key columns
    float Mrow = rs[m][0];
    float invL = 1.0f / rs[m][1];
    size_t srow = (((size_t)(b * NH + h) * SS) + (i0 + m)) * SS + j0;

    v16bf pa;
#pragma unroll
    for (int e = 0; e < 8; ++e) {
      float s = probs[srow + kb0 + e];
      float p = __expf(s - Mrow) * invL;
      probs[srow + kb0 + e] = p;
      pa[e] = (bf16_t)p;
    }
#pragma unroll
    for (int e = 0; e < 8; ++e) {
      float s = probs[srow + kb0 + 16 + e];
      float p = __expf(s - Mrow) * invL;
      probs[srow + kb0 + 16 + e] = p;
      pa[8 + e] = (bf16_t)p;
    }

#pragma unroll
    for (int ni = 0; ni < 4; ++ni) {
      // B[k][n] = V[j0+k][ni*16+n] = vst[(ni*16+n)*LDJ + w*32 + k]
      v16bf bv = load_frag_bt(&vst[(ni * 16) * LDJ + w * 32], LDJ, lane);
      O[ni] = wmma_bf16(pa, bv, O[ni]);
    }
  }

  // Deterministic cross-wave reduction of the 16x64 ctx tile (reuse vst).
  __syncthreads();
  float* red = (float*)vst;          // 8*16*64*4 = 32768 B <= 33792
  int nlane = lane & 15, mbv = (lane < 16) ? 0 : 8;
#pragma unroll
  for (int ni = 0; ni < 4; ++ni)
#pragma unroll
    for (int r = 0; r < 8; ++r)
      red[(w * 16 + mbv + r) * 64 + ni * 16 + nlane] = O[ni][r];
  __syncthreads();
#pragma unroll
  for (int it = 0; it < 4; ++it) {
    int t = tid + it * 256;
    int r = t >> 6, d = t & 63;
    float s = 0.f;
#pragma unroll
    for (int ww = 0; ww < 8; ++ww) s += red[(ww * 16 + r) * 64 + d];
    ctx[((size_t)(b * SS + i0 + r)) * HID + h * HD + d] = (bf16_t)s;
  }
}

// ---------------------------------------------------------------------------
// Kernel 4a: y = ctx @ Wop + bop + hidden   (fp32 into d_out)
// ---------------------------------------------------------------------------
__global__ __launch_bounds__(256) void out_gemm(
    const bf16_t* __restrict__ ctxb, const bf16_t* __restrict__ Wopb,
    const float* __restrict__ bo, const float* __restrict__ gamma_p,
    const float* __restrict__ hidden, float* __restrict__ y)
{
  const int LDA = 40, LDK = 40;
  __shared__ __align__(16) bf16_t As[128 * LDA];
  __shared__ __align__(16) bf16_t Bt[64 * LDK];

  int m0 = blockIdx.x * 128, n0 = blockIdx.y * 64;
  int tid = threadIdx.x, lane = tid & 31, w = tid >> 5;
  int wm = w & 3, wn = w >> 2;
  v8f acc[2][2] = {{{0}}};

  for (int k0 = 0; k0 < HID; k0 += 32) {
#pragma unroll
    for (int it = 0; it < 2; ++it) {
      int t = tid + it * 256;
      int r = t >> 2, c8 = (t & 3) * 8;
      *(u32x4*)&As[r * LDA + c8] =
          *(const u32x4*)&ctxb[(size_t)(m0 + r) * HID + k0 + c8];
    }
#pragma unroll
    for (int it = 0; it < 2; ++it) {
      int t = tid + it * 256;
      int n = t & 63, k4 = (t >> 6) * 4;
      union { unsigned long long u; bf16_t h[4]; } pk;
#pragma unroll
      for (int j = 0; j < 4; ++j)
        pk.h[j] = Wopb[(size_t)(k0 + k4 + j) * HID + n0 + n];
      *(unsigned long long*)&Bt[n * LDK + k4] = pk.u;
    }
    if (k0 + 32 < HID)
      __builtin_prefetch(&ctxb[(size_t)(m0 + (tid >> 1)) * HID + k0 + 32], 0, 0);
    __syncthreads();

    v16bf afr[2];
    afr[0] = load_frag_a(&As[(wm * 32 + 0)  * LDA], LDA, lane);
    afr[1] = load_frag_a(&As[(wm * 32 + 16) * LDA], LDA, lane);
#pragma unroll
    for (int ni = 0; ni < 2; ++ni) {
      v16bf bfr = load_frag_bt(&Bt[(wn * 32 + ni * 16) * LDK], LDK, lane);
#pragma unroll
      for (int mi = 0; mi < 2; ++mi)
        acc[mi][ni] = wmma_bf16(afr[mi], bfr, acc[mi][ni]);
    }
    __syncthreads();
  }

  float g = *gamma_p;
  int nlane = lane & 15, mb = (lane < 16) ? 0 : 8;
#pragma unroll
  for (int mi = 0; mi < 2; ++mi)
#pragma unroll
    for (int ni = 0; ni < 2; ++ni)
#pragma unroll
      for (int r = 0; r < 8; ++r) {
        int row = m0 + wm * 32 + mi * 16 + mb + r;
        int col = n0 + wn * 32 + ni * 16 + nlane;
        float bv = bo[col];
        float bop = bv + g * fmaxf(bv, 0.f);
        y[(size_t)row * HID + col] =
            acc[mi][ni][r] + bop + hidden[(size_t)row * HID + col];
      }
}

// ---------------------------------------------------------------------------
// Kernel 4b: in-place rowwise LayerNorm (no weight/bias), eps = 1e-12
// ---------------------------------------------------------------------------
__global__ __launch_bounds__(256) void layernorm_inplace(float* __restrict__ y) {
  __shared__ float r1[256], r2[256];
  int row = blockIdx.x, tid = threadIdx.x;
  float* p = y + (size_t)row * HID;
  float vals[4], s = 0.f, ss = 0.f;
#pragma unroll
  for (int i = 0; i < 4; ++i) {
    float v = p[tid + i * 256];
    vals[i] = v; s += v; ss += v * v;
  }
  r1[tid] = s; r2[tid] = ss;
  __syncthreads();
  for (int off = 128; off > 0; off >>= 1) {
    if (tid < off) { r1[tid] += r1[tid + off]; r2[tid] += r2[tid + off]; }
    __syncthreads();
  }
  float mu = r1[0] * (1.0f / HID);
  float var = r2[0] * (1.0f / HID) - mu * mu;
  float rstd = rsqrtf(var + EPS_LN);
#pragma unroll
  for (int i = 0; i < 4; ++i) p[tid + i * 256] = (vals[i] - mu) * rstd;
}

// ---------------------------------------------------------------------------
// Launch
// ---------------------------------------------------------------------------
extern "C" void kernel_launch(void* const* d_in, const int* in_sizes, int n_in,
                              void* d_out, int out_size, void* d_ws, size_t ws_size,
                              hipStream_t stream) {
  const float* hidden = (const float*)d_in[0];
  const float* Wq = (const float*)d_in[1];
  const float* bq = (const float*)d_in[2];
  const float* Wk = (const float*)d_in[3];
  const float* bk = (const float*)d_in[4];
  const float* Wv = (const float*)d_in[5];
  const float* bv = (const float*)d_in[6];
  const float* Wo = (const float*)d_in[7];
  const float* bo = (const float*)d_in[8];
  const float* gamma = (const float*)d_in[9];

  char* ws = (char*)d_ws;
  const size_t MB = 1ull << 20;
  bf16_t* xb    = (bf16_t*)(ws + 0 * MB);   // 8 MB
  bf16_t* Wqb   = (bf16_t*)(ws + 8 * MB);   // 2 MB each
  bf16_t* Wqpb  = (bf16_t*)(ws + 10 * MB);
  bf16_t* Wkb   = (bf16_t*)(ws + 12 * MB);
  bf16_t* Wkpb  = (bf16_t*)(ws + 14 * MB);
  bf16_t* Wvb   = (bf16_t*)(ws + 16 * MB);
  bf16_t* Wvpb  = (bf16_t*)(ws + 18 * MB);
  bf16_t* Wopb  = (bf16_t*)(ws + 20 * MB);
  bf16_t* qbuf  = (bf16_t*)(ws + 22 * MB);  // 8 MB each
  bf16_t* kbuf  = (bf16_t*)(ws + 30 * MB);
  bf16_t* vbuf  = (bf16_t*)(ws + 38 * MB);
  bf16_t* ctxb  = (bf16_t*)(ws + 46 * MB);
  float*  rowstats = (float*)(ws + 54 * MB); // 512 KB

  float* outy  = (float*)d_out;                        // [2,2048,1024]
  float* probs = (float*)d_out + (size_t)MROWS * HID;  // [2,16,2048,2048]

  // 1) precision prep
  prep_convert<<<4096, 256, 0, stream>>>(hidden, xb, nullptr, gamma, MROWS * HID);
  prep_convert<<<1024, 256, 0, stream>>>(Wq, Wqb, Wqpb, gamma, HID * HID);
  prep_convert<<<1024, 256, 0, stream>>>(Wk, Wkb, Wkpb, gamma, HID * HID);
  prep_convert<<<1024, 256, 0, stream>>>(Wv, Wvb, Wvpb, gamma, HID * HID);
  prep_convert<<<1024, 256, 0, stream>>>(Wo, nullptr, Wopb, gamma, HID * HID);

  // 2) QKV projections with gamma-rule combine
  dim3 ggrid(MROWS / 128, HID / 64);
  qkv_gemm<<<ggrid, 256, 0, stream>>>(xb, Wqb, Wqpb, bq, gamma, qbuf);
  qkv_gemm<<<ggrid, 256, 0, stream>>>(xb, Wkb, Wkpb, bk, gamma, kbuf);
  qkv_gemm<<<ggrid, 256, 0, stream>>>(xb, Wvb, Wvpb, bv, gamma, vbuf);

  // 3) attention: two-pass softmax with probs materialized in d_out
  int ablocks = BB * NH * (SS / 16);  // 4096
  attn_scores<<<ablocks, 256, 0, stream>>>(qbuf, kbuf, probs, rowstats);
  attn_ctx<<<ablocks, 256, 0, stream>>>(vbuf, probs, rowstats, ctxb);

  // 4) output projection + residual, then LayerNorm in place
  out_gemm<<<ggrid, 256, 0, stream>>>(ctxb, Wopb, bo, gamma, hidden, outy);
  layernorm_inplace<<<MROWS, 256, 0, stream>>>(outy);
}
